// neko_lens_self_20100446945742
// MI455X (gfx1250) — matrix operand order repaired and checked by reading, not compiled, not run on hardware
//
#include <hip/hip_runtime.h>
#include <hip/hip_bf16.h>
#include <math.h>

// ---------------------------------------------------------------------------
// Problem constants: feat (16,64,256,256) f32, conv1 3x3 pad1 -> BN -> ReLU6
// -> channel L2-norm -> conv2 5x5 valid (2ch) -> sigmoid -> resize chain ->
// cumsum grid -> grid_sample.  Outputs: dst (16,64,128,128) + dmap (16,2,130,130)
// ---------------------------------------------------------------------------

typedef __attribute__((ext_vector_type(16))) __bf16 v16bf;
typedef __attribute__((ext_vector_type(8)))  __bf16 v8bf;
typedef __attribute__((ext_vector_type(8)))  float  v8f;

#define B_N      16
#define CH       64
#define H_IN     256
#define W_IN     256
#define HW_IN    65536           // 256*256

// ---------------------------------------------------------------------------
// K0: fold conv bias + BN into per-channel scale/shift
//   y = (conv + b1 - rmean) * gamma*rsqrt(var+eps) + beta = conv*sc + sh
// ---------------------------------------------------------------------------
__global__ void k0_bnfold(const float* __restrict__ b1, const float* __restrict__ gamma,
                          const float* __restrict__ beta, const float* __restrict__ rmean,
                          const float* __restrict__ rvar, float* __restrict__ sc,
                          float* __restrict__ sh) {
  int c = threadIdx.x;
  if (c < CH) {
    float s = gamma[c] * rsqrtf(rvar[c] + 1e-5f);
    sc[c] = s;
    sh[c] = (b1[c] - rmean[c]) * s + beta[c];
  }
}

// ---------------------------------------------------------------------------
// K1: conv1 as implicit GEMM with WMMA bf16 (M=64, N=128-pixel tile, K=576).
// K reordered tap-major: k = tap*64 + ci, so each K=32 chunk is one 3x3 tap
// and 32 contiguous input channels.  Fuses BN + ReLU6 + cross-channel L2 norm.
// Block: 256 threads = 8 waves laid out 4(M) x 2(N).  Grid: (512 n-tiles, 16 b).
// ---------------------------------------------------------------------------
__global__ __launch_bounds__(256)
void k1_conv1_wmma(const float* __restrict__ feat, const float* __restrict__ w1,
                   const float* __restrict__ sc, const float* __restrict__ sh,
                   float* __restrict__ xn) {
  __shared__ __align__(16) __bf16 Al[64 * 40];    // A tile: 64 m x 32 k (pad 40)
  __shared__ __align__(16) __bf16 Bl[128 * 40];   // B tile: 128 n x 32 k (pad 40)
  __shared__ float ssum[128];                     // per-pixel sum of squares

  const int tid = threadIdx.x;
  const int b  = blockIdx.y;
  const int n0 = blockIdx.x * 128;                // pixel tile base (row-major)
  const int h  = n0 >> 8;                         // single image row per tile
  const int w0 = n0 & 255;                        // 0 or 128

  if (tid < 128) ssum[tid] = 0.0f;

  const int lane = tid & 31;
  const int wid  = tid >> 5;
  const int wm   = wid & 3;                       // which 16-row M block
  const int wn   = wid >> 2;                      // which 64-pixel N half
  const int m0   = wm * 16;
  const int lrow = lane & 15;
  const bool hiL = lane >= 16;

  v8f acc[4] = {};                                // 4 x (16x16) C tiles

  const float* featb = feat + (size_t)b * CH * HW_IN;

  for (int s = 0; s < 18; ++s) {                  // 18 K-steps of 32
    const int tap = s >> 1;                       // 0..8  (kh*3+kw)
    const int ci0 = (s & 1) * 32;
    const int dh = tap / 3 - 1;
    const int dw = tap % 3 - 1;

    __syncthreads();                              // previous tile consumed

    // ---- stage B: im2col, 32 k x 128 n, f32 -> bf16 --------------------
    {
      const int wl  = tid & 127;
      const int kk0 = (tid >> 7) * 16;
      const int hh  = h + dh;
      const int ww  = w0 + wl + dw;
      const bool ok = ((unsigned)hh < (unsigned)H_IN) && ((unsigned)ww < (unsigned)W_IN);
      #pragma unroll
      for (int i = 0; i < 16; ++i) {
        const int kk = kk0 + i;
        float v = 0.0f;
        if (ok) v = featb[(size_t)(ci0 + kk) * HW_IN + (hh << 8) + ww];
        Bl[wl * 40 + kk] = (__bf16)v;
      }
    }
    // ---- stage A: weights, 64 m x 32 k ---------------------------------
    #pragma unroll
    for (int i = 0; i < 8; ++i) {
      const int idx = tid * 8 + i;
      const int m  = idx >> 5;
      const int kk = idx & 31;
      Al[m * 40 + kk] = (__bf16)w1[(size_t)(m * CH + ci0 + kk) * 9 + tap];
    }
    __syncthreads();

    // ---- A fragment (16-bit A layout: lanes<16 K 0..7,16..23; else +8) --
    const int kb = hiL ? 8 : 0;
    const __bf16* ap = Al + (m0 + lrow) * 40;
    v8bf alo = *(const v8bf*)(ap + kb);
    v8bf ahi = *(const v8bf*)(ap + 16 + kb);
    v16bf afrag = __builtin_shufflevector(alo, ahi, 0,1,2,3,4,5,6,7,8,9,10,11,12,13,14,15);

    // ---- B fragments + WMMA --------------------------------------------
    const int koff = hiL ? 8 : 0;                 // lanes<16: K 0..7,16..23; else +8
    #pragma unroll
    for (int nt = 0; nt < 4; ++nt) {
      const int nrow = wn * 64 + nt * 16 + lrow;
      const __bf16* bp = Bl + nrow * 40 + koff;
      v8bf blo = *(const v8bf*)(bp);
      v8bf bhi = *(const v8bf*)(bp + 16);
      v16bf bfrag = __builtin_shufflevector(blo, bhi, 0,1,2,3,4,5,6,7,8,9,10,11,12,13,14,15);
      acc[nt] = __builtin_amdgcn_wmma_f32_16x16x32_bf16(
          false, afrag, false, bfrag, (short)0, acc[nt], false, false);
    }
  }

  // ---- epilogue: BN + ReLU6, then cross-channel sum of squares -----------
  #pragma unroll
  for (int nt = 0; nt < 4; ++nt) {
    float local = 0.0f;
    #pragma unroll
    for (int r = 0; r < 8; ++r) {
      const int co = m0 + (hiL ? 8 + r : r);
      float v = acc[nt][r] * sc[co] + sh[co];
      v = fminf(fmaxf(v, 0.0f), 6.0f);
      acc[nt][r] = v;
      local += v * v;
    }
    float tot = local + __shfl_xor(local, 16, 32);  // combine M halves per pixel
    if (!hiL) atomicAdd(&ssum[wn * 64 + nt * 16 + lrow], tot);
  }
  __syncthreads();

  // ---- normalize and store xn --------------------------------------------
  #pragma unroll
  for (int nt = 0; nt < 4; ++nt) {
    const int nl = wn * 64 + nt * 16 + lrow;
    const float nrm = sqrtf(ssum[nl]);
    const float rs  = (nrm > 0.09f) ? 1.0f / (nrm + 9e-8f) : 0.0f;
    #pragma unroll
    for (int r = 0; r < 8; ++r) {
      const int co = m0 + (hiL ? 8 + r : r);
      xn[((size_t)(b * CH + co) << 16) + n0 + nl] = acc[nt][r] * rs;
    }
  }
}

// ---------------------------------------------------------------------------
// K2: conv2 5x5 valid (64 -> 2 channels) + sigmoid.  LDS-tiled direct conv.
// Block 256 = 16x16 output tile.  Grid (16,16,16).
// ---------------------------------------------------------------------------
__global__ __launch_bounds__(256)
void k2_conv2_sig(const float* __restrict__ xn, const float* __restrict__ w2,
                  const float* __restrict__ b2, float* __restrict__ draw) {
  __shared__ float wl[2 * CH * 25];     // 3200 floats
  __shared__ float til[20 * 20];

  const int tid = threadIdx.x;
  const int b   = blockIdx.z;
  const int ho0 = blockIdx.y * 16;
  const int wo0 = blockIdx.x * 16;
  const int tx = tid & 15, ty = tid >> 4;

  for (int i = tid; i < 2 * CH * 25; i += 256) wl[i] = w2[i];

  float a0 = b2[0], a1 = b2[1];
  const float* xb = xn + (size_t)b * CH * HW_IN;

  for (int ci = 0; ci < CH; ++ci) {
    __syncthreads();
    for (int i = tid; i < 400; i += 256) {
      const int r = i / 20, c2 = i % 20;
      const int hh = ho0 + r, ww = wo0 + c2;
      til[i] = (hh < H_IN && ww < W_IN) ? xb[(size_t)ci * HW_IN + (hh << 8) + ww] : 0.0f;
    }
    __syncthreads();
    const float* wp0 = wl + ci * 25;
    const float* wp1 = wl + CH * 25 + ci * 25;
    #pragma unroll
    for (int kh = 0; kh < 5; ++kh) {
      #pragma unroll
      for (int kw = 0; kw < 5; ++kw) {
        const float v = til[(ty + kh) * 20 + tx + kw];
        a0 = fmaf(v, wp0[kh * 5 + kw], a0);
        a1 = fmaf(v, wp1[kh * 5 + kw], a1);
      }
    }
  }
  const int ho = ho0 + ty, wo = wo0 + tx;
  if (ho < 252 && wo < 252) {
    const size_t o0 = ((size_t)(b * 2 + 0) * 252 + ho) * 252 + wo;
    const size_t o1 = ((size_t)(b * 2 + 1) * 252 + ho) * 252 + wo;
    draw[o0] = 1.0f / (1.0f + __expf(-a0));
    draw[o1] = 1.0f / (1.0f + __expf(-a1));
  }
}

// ---------------------------------------------------------------------------
// K3a: nearest resize 252 -> 130 + 0.0009 + hardness mask
// ---------------------------------------------------------------------------
__global__ void k3a_nearest_mask(const float* __restrict__ draw, float* __restrict__ dmap0) {
  const int idx = blockIdx.x * 256 + threadIdx.x;
  const int total = B_N * 2 * 130 * 130;
  if (idx >= total) return;
  const int j  = idx % 130;
  const int t1 = idx / 130;
  const int i  = t1 % 130;
  const int bc = t1 / 130;
  const int hi_ = (int)floorf((float)i * (252.0f / 130.0f));
  const int wi_ = (int)floorf((float)j * (252.0f / 130.0f));
  const float v = draw[(size_t)bc * 252 * 252 + hi_ * 252 + wi_];
  const float m = (i >= 1 && i <= 128 && j >= 1 && j <= 128) ? 2.0f : 0.0f;
  dmap0[idx] = v + 0.0009f + m;
}

// ---------------------------------------------------------------------------
// Generic 2D bilinear resize (half-pixel centers, clamped), per (b,ch) plane.
// ---------------------------------------------------------------------------
__global__ void k_bilinear(const float* __restrict__ src, float* __restrict__ dst,
                           int IH, int IW, int OH, int OW, int total) {
  const int idx = blockIdx.x * 256 + threadIdx.x;
  if (idx >= total) return;
  const int ox = idx % OW;
  const int t1 = idx / OW;
  const int oy = t1 % OH;
  const int bc = t1 / OH;
  const float fy = ((float)oy + 0.5f) * ((float)IH / (float)OH) - 0.5f;
  const float fx = ((float)ox + 0.5f) * ((float)IW / (float)OW) - 0.5f;
  const float y0f = floorf(fy), x0f = floorf(fx);
  const float wy = fy - y0f, wx = fx - x0f;
  int y0 = (int)y0f, x0 = (int)x0f;
  int y1 = y0 + 1, x1 = x0 + 1;
  y0 = min(max(y0, 0), IH - 1); y1 = min(max(y1, 0), IH - 1);
  x0 = min(max(x0, 0), IW - 1); x1 = min(max(x1, 0), IW - 1);
  const float* p = src + (size_t)bc * IH * IW;
  const float v00 = p[y0 * IW + x0], v01 = p[y0 * IW + x1];
  const float v10 = p[y1 * IW + x0], v11 = p[y1 * IW + x1];
  dst[idx] = v00 * (1.0f - wx) * (1.0f - wy) + v01 * wx * (1.0f - wy)
           + v10 * (1.0f - wx) * wy + v11 * wx * wy;
}

// ---------------------------------------------------------------------------
// K4h: per-(b,col) normalize + cumsum of dmap channel 0 (rows)  -> hc
// K4w: per-(b,row) normalize + cumsum of dmap channel 1 (cols)  -> wc
// ---------------------------------------------------------------------------
__global__ void k4_hcum(const float* __restrict__ dmap, float* __restrict__ hc) {
  const int idx = blockIdx.x * 128 + threadIdx.x;
  if (idx >= B_N * 130) return;
  const int b = idx / 130, c = idx % 130;
  const float* p = dmap + (size_t)(b * 2 + 0) * 130 * 130 + c;
  float sum = 0.0f;
  for (int r = 0; r < 130; ++r) sum += p[r * 130];
  const float inv = 1.0f / sum;
  float run = 0.0f;
  for (int r = 0; r < 130; ++r) {
    run += p[r * 130];
    hc[((size_t)b * 130 + r) * 130 + c] = run * inv;
  }
}

__global__ void k4_wcum(const float* __restrict__ dmap, float* __restrict__ wc) {
  const int idx = blockIdx.x * 128 + threadIdx.x;
  if (idx >= B_N * 130) return;
  const int b = idx / 130, r = idx % 130;
  const float* p = dmap + ((size_t)(b * 2 + 1) * 130 + r) * 130;
  float sum = 0.0f;
  for (int c = 0; c < 130; ++c) sum += p[c];
  const float inv = 1.0f / sum;
  float run = 0.0f;
  for (int c = 0; c < 130; ++c) {
    run += p[c];
    wc[((size_t)b * 130 + r) * 130 + c] = run * inv;
  }
}

// ---------------------------------------------------------------------------
// K5: grid_sample bilinear.  Block = 128 x-threads, grid (128 y, 16 b).
// Per-pixel corner weights computed once, reused across all 64 channels.
// ---------------------------------------------------------------------------
__global__ __launch_bounds__(128)
void k5_grid_sample(const float* __restrict__ feat, const float* __restrict__ hc,
                    const float* __restrict__ wc, float* __restrict__ dst) {
  const int x = threadIdx.x;
  const int y = blockIdx.x;
  const int b = blockIdx.y;

  const float gx = wc[((size_t)b * 130 + (y + 1)) * 130 + (x + 1)] * 2.0f - 1.0f;
  const float gy = hc[((size_t)b * 130 + (y + 1)) * 130 + (x + 1)] * 2.0f - 1.0f;
  const float ix = (gx + 1.0f) * ((float)W_IN * 0.5f) - 0.5f;
  const float iy = (gy + 1.0f) * ((float)H_IN * 0.5f) - 0.5f;
  const float x0f = floorf(ix), y0f = floorf(iy);
  const float wx = ix - x0f, wy = iy - y0f;
  const int x0 = (int)x0f, y0 = (int)y0f;
  const int x1 = x0 + 1, y1 = y0 + 1;

  const float m00 = ((unsigned)x0 < (unsigned)W_IN && (unsigned)y0 < (unsigned)H_IN) ? 1.0f : 0.0f;
  const float m01 = ((unsigned)x1 < (unsigned)W_IN && (unsigned)y0 < (unsigned)H_IN) ? 1.0f : 0.0f;
  const float m10 = ((unsigned)x0 < (unsigned)W_IN && (unsigned)y1 < (unsigned)H_IN) ? 1.0f : 0.0f;
  const float m11 = ((unsigned)x1 < (unsigned)W_IN && (unsigned)y1 < (unsigned)H_IN) ? 1.0f : 0.0f;
  const int x0c = min(max(x0, 0), W_IN - 1), x1c = min(max(x1, 0), W_IN - 1);
  const int y0c = min(max(y0, 0), H_IN - 1), y1c = min(max(y1, 0), H_IN - 1);
  const float w00 = (1.0f - wx) * (1.0f - wy) * m00;
  const float w01 = wx * (1.0f - wy) * m01;
  const float w10 = (1.0f - wx) * wy * m10;
  const float w11 = wx * wy * m11;
  const int o00 = y0c * W_IN + x0c, o01 = y0c * W_IN + x1c;
  const int o10 = y1c * W_IN + x0c, o11 = y1c * W_IN + x1c;

  const float* fb = feat + (size_t)b * CH * HW_IN;
  for (int c = 0; c < CH; ++c) {
    const float* fc = fb + (size_t)c * HW_IN;
    const float v = fc[o00] * w00 + fc[o01] * w01 + fc[o10] * w10 + fc[o11] * w11;
    dst[(((size_t)(b * CH + c)) * 128 + y) * 128 + x] = v;
  }
}

// ---------------------------------------------------------------------------
// Launch sequence
// ---------------------------------------------------------------------------
extern "C" void kernel_launch(void* const* d_in, const int* in_sizes, int n_in,
                              void* d_out, int out_size, void* d_ws, size_t ws_size,
                              hipStream_t stream) {
  const float* feat  = (const float*)d_in[0];
  const float* w1    = (const float*)d_in[1];
  const float* b1    = (const float*)d_in[2];
  const float* gamma = (const float*)d_in[3];
  const float* beta  = (const float*)d_in[4];
  const float* rmean = (const float*)d_in[5];
  const float* rvar  = (const float*)d_in[6];
  const float* w2    = (const float*)d_in[7];
  const float* b2    = (const float*)d_in[8];

  float* ws    = (float*)d_ws;
  float* sc    = ws;                       // 64
  float* sh    = ws + 64;                  // 64
  float* xn    = ws + 128;                 // 16*64*256*256 = 67,108,864
  float* draw  = xn + (size_t)67108864;    // 16*2*252*252  =  2,032,128
  float* dmap0 = draw + (size_t)2032128;   // 16*2*130*130  =    540,800
  float* dsm   = dmap0 + (size_t)540800;   // 16*2*128*128  =    524,288
  float* hc    = dsm + (size_t)524288;     // 16*130*130    =    270,400
  float* wcb   = hc + (size_t)270400;      // 16*130*130    =    270,400

  float* dst     = (float*)d_out;                 // 16*64*128*128
  float* dmapout = dst + (size_t)16777216;        // 16*2*130*130

  k0_bnfold<<<1, 64, 0, stream>>>(b1, gamma, beta, rmean, rvar, sc, sh);

  k1_conv1_wmma<<<dim3(512, 16), 256, 0, stream>>>(feat, w1, sc, sh, xn);

  k2_conv2_sig<<<dim3(16, 16, 16), 256, 0, stream>>>(xn, w2, b2, draw);

  k3a_nearest_mask<<<(16 * 2 * 130 * 130 + 255) / 256, 256, 0, stream>>>(draw, dmap0);

  k_bilinear<<<(16 * 2 * 128 * 128 + 255) / 256, 256, 0, stream>>>(
      dmap0, dsm, 130, 130, 128, 128, 16 * 2 * 128 * 128);

  k_bilinear<<<(16 * 2 * 130 * 130 + 255) / 256, 256, 0, stream>>>(
      dsm, dmapout, 128, 128, 130, 130, 16 * 2 * 130 * 130);

  k4_hcum<<<(16 * 130 + 127) / 128, 128, 0, stream>>>(dmapout, hc);
  k4_wcum<<<(16 * 130 + 127) / 128, 128, 0, stream>>>(dmapout, wcb);

  k5_grid_sample<<<dim3(128, 16), 128, 0, stream>>>(feat, hc, wcb, dst);
}
